// DentalBoundaryDGCNN_25340307046482
// MI455X (gfx1250) — matrix operand order repaired and checked by reading, not compiled, not run on hardware
//
#include <hip/hip_runtime.h>
#include <cstddef>
#include <cstdint>

// ---------------------------------------------------------------------------
// DentalBoundaryDGCNN forward for MI455X (gfx1250, wave32, WMMA bf16->f32)
// B=4 graphs, P=4096 pts, K=20 edge-knn, KB=10 boundary-knn
// ---------------------------------------------------------------------------

typedef __attribute__((ext_vector_type(16))) __bf16 v16bf;
typedef __attribute__((ext_vector_type(8)))  float  v8f;

#define BATCH 4
#define PTS   4096
#define NPT   16384      // BATCH*PTS
#define KEDGE 20
#define KBND  10

// ------------------------------- helpers -----------------------------------

__device__ __forceinline__ unsigned short f2bf(float x) {
  unsigned u = __float_as_uint(x);
  unsigned r = u + 0x7FFFu + ((u >> 16) & 1u);   // round-to-nearest-even
  return (unsigned short)(r >> 16);
}
__device__ __forceinline__ float bf2f(unsigned short h) {
  return __uint_as_float(((unsigned)h) << 16);
}
__device__ __forceinline__ float wred(float v) {   // wave32 sum reduction
#pragma unroll
  for (int o = 16; o > 0; o >>= 1) v += __shfl_xor(v, o, 32);
  return v;
}

// ----------------------- conversion / small kernels ------------------------

// fp32 [rows, lds_] (use first C cols) -> bf16 [rows, ldd] zero-padded
__global__ __launch_bounds__(256) void k_cvt(const float* __restrict__ src, int lds_, int C,
                                             unsigned short* __restrict__ dst, int ldd, int rows) {
  long long tid = (long long)blockIdx.x * 256 + threadIdx.x;
  if (tid >= (long long)rows * ldd) return;
  int r = (int)(tid / ldd), c = (int)(tid % ldd);
  float v = (c < C) ? src[(size_t)r * lds_ + c] : 0.f;
  dst[(size_t)r * ldd + c] = f2bf(v);
}

// weight fp32 [N,K] -> bf16 [N,Kpad] zero-padded
__global__ __launch_bounds__(256) void k_cvtw(const float* __restrict__ w, int K, int Kpad,
                                              unsigned short* __restrict__ dst, int N) {
  long long tid = (long long)blockIdx.x * 256 + threadIdx.x;
  if (tid >= (long long)N * Kpad) return;
  int n = (int)(tid / Kpad), k = (int)(tid % Kpad);
  float v = (k < K) ? w[(size_t)n * K + k] : 0.f;
  dst[(size_t)n * Kpad + k] = f2bf(v);
}

// row squared-norm of bf16 feature matrix
__global__ __launch_bounds__(256) void k_sqnorm(const unsigned short* __restrict__ f, int ld, int C,
                                                float* __restrict__ d2, int rows) {
  int r = blockIdx.x * 256 + threadIdx.x;
  if (r >= rows) return;
  const unsigned short* p = f + (size_t)r * ld;
  float s = 0.f;
  for (int c = 0; c < C; ++c) { float v = bf2f(p[c]); s += v * v; }
  d2[r] = s;
}

// ------------------------- WMMA GEMM (bf16 -> f32) -------------------------
// out[M,N] = A[M,KDIM](bf16,row-major,lda) * W[N,KDIM](bf16,row-major)^T + bias
// block = 256 (8 waves); wave -> 16x64 tile; grid = (M/128, N/64)
// KDIM is a template constant: the ping-pong double buffer fully unrolls, so
// no register-rotation copies and all addresses fold into load offsets.

union Frag { v16bf v; unsigned u[8]; };

__device__ __forceinline__ void load_afrag(const unsigned short* Ar, int kk, int kbA, Frag& a) {
#pragma unroll
  for (int e = 0; e < 4; ++e) {
    a.u[e]     = *(const unsigned*)(Ar + kk + kbA + 2 * e);        // K = kb+0..7
    a.u[4 + e] = *(const unsigned*)(Ar + kk + 16 + kbA + 2 * e);   // K = 16+kb+0..7
  }
}
__device__ __forceinline__ void load_bfrag(const unsigned short* Bt, int kk, int kbB, Frag& b) {
#pragma unroll
  for (int e = 0; e < 8; ++e) b.u[e] = *(const unsigned*)(Bt + kk + kbB + (e >> 2) * 16 + 2 * (e & 3));
}

template <int KDIM>
__global__ __launch_bounds__(256) void k_gemm(const unsigned short* __restrict__ A, int lda,
                                              const unsigned short* __restrict__ W, int ldb,
                                              const float* __restrict__ bias,
                                              float* __restrict__ out, int ldc, int M) {
  constexpr int NS = KDIM / 32;
  int lane = threadIdx.x & 31, wave = threadIdx.x >> 5;
  int m0 = (blockIdx.x * 8 + wave) * 16;
  int n0 = blockIdx.y * 64;
  if (m0 >= M) return;
  int mrow = m0 + (lane & 15);
  int kbA  = (lane >> 4) * 8;
  int ncol = lane & 15;
  int kbB  = (lane >> 4) * 8;
  const unsigned short* Ar = A + (size_t)mrow * lda;
  const unsigned short* Bp[4];
#pragma unroll
  for (int t = 0; t < 4; ++t) Bp[t] = W + (size_t)(n0 + 16 * t + ncol) * ldb;
  v8f acc[4] = {};
  Frag a[2], b[2][4];
  load_afrag(Ar, 0, kbA, a[0]);
#pragma unroll
  for (int t = 0; t < 4; ++t) load_bfrag(Bp[t], 0, kbB, b[0][t]);
#pragma unroll
  for (int s = 0; s < NS; ++s) {
    const int cur = s & 1, nxt = cur ^ 1;
    if (s + 2 < NS) __builtin_prefetch(Ar + (s + 2) * 32, 0, 1);
    if (s + 1 < NS) {
      load_afrag(Ar, (s + 1) * 32, kbA, a[nxt]);
#pragma unroll
      for (int t = 0; t < 4; ++t) load_bfrag(Bp[t], (s + 1) * 32, kbB, b[nxt][t]);
    }
#pragma unroll
    for (int t = 0; t < 4; ++t)
      acc[t] = __builtin_amdgcn_wmma_f32_16x16x32_bf16(false, a[cur].v, false, b[cur][t].v,
                                                       (short)0, acc[t], false, false);
  }
  int mbase = m0 + (lane >> 4) * 8;
#pragma unroll
  for (int v = 0; v < 8; ++v) {
    int m = mbase + v;
    float* o = out + (size_t)m * ldc + n0 + ncol;
#pragma unroll
    for (int t = 0; t < 4; ++t) o[16 * t] = acc[t][v] + bias[n0 + 16 * t + ncol];
  }
}

// pairwise distance: dist[m][n] = d2[m] + d2[n] - 2 * <F_m, F_n>   (P x P)
template <int KDIM>
__global__ __launch_bounds__(256) void k_dist(const unsigned short* __restrict__ F, int ld,
                                              const float* __restrict__ d2,
                                              float* __restrict__ dist) {
  constexpr int NS = KDIM / 32;
  int lane = threadIdx.x & 31, wave = threadIdx.x >> 5;
  int m0 = (blockIdx.x * 8 + wave) * 16;
  int n0 = blockIdx.y * 64;
  int mrow = m0 + (lane & 15);
  int kbA  = (lane >> 4) * 8;
  int ncol = lane & 15;
  int kbB  = (lane >> 4) * 8;
  const unsigned short* Ar = F + (size_t)mrow * ld;
  const unsigned short* Bp[4];
#pragma unroll
  for (int t = 0; t < 4; ++t) Bp[t] = F + (size_t)(n0 + 16 * t + ncol) * ld;
  v8f acc[4] = {};
  Frag a[2], b[2][4];
  load_afrag(Ar, 0, kbA, a[0]);
#pragma unroll
  for (int t = 0; t < 4; ++t) load_bfrag(Bp[t], 0, kbB, b[0][t]);
#pragma unroll
  for (int s = 0; s < NS; ++s) {
    const int cur = s & 1, nxt = cur ^ 1;
    if (s + 1 < NS) {
      load_afrag(Ar, (s + 1) * 32, kbA, a[nxt]);
#pragma unroll
      for (int t = 0; t < 4; ++t) load_bfrag(Bp[t], (s + 1) * 32, kbB, b[nxt][t]);
    }
#pragma unroll
    for (int t = 0; t < 4; ++t)
      acc[t] = __builtin_amdgcn_wmma_f32_16x16x32_bf16(false, a[cur].v, false, b[cur][t].v,
                                                       (short)0, acc[t], false, false);
  }
  int mbase = m0 + (lane >> 4) * 8;
#pragma unroll
  for (int v = 0; v < 8; ++v) {
    int m = mbase + v;
    float dm = d2[m];
#pragma unroll
    for (int t = 0; t < 4; ++t) {
      int n = n0 + 16 * t + ncol;
      dist[(size_t)m * PTS + n] = dm + d2[n] - 2.f * acc[t][v];
    }
  }
}

// --------------------------- top-k (k smallest) ----------------------------
// one block (256 thr) per row; row staged into LDS with CDNA5 async
// global->LDS copies (ASYNCcnt), then k min-extractions in LDS.
__global__ __launch_bounds__(256) void k_topk(const float* __restrict__ dist,
                                              int* __restrict__ idx, int n, int k) {
  __shared__ __align__(16) float s[PTS];
  __shared__ float rv[256];
  __shared__ int   ri[256];
  int row = blockIdx.x, t = threadIdx.x;
  const float* d = dist + (size_t)row * n;
  // async B128 copies: 16 bytes per op, 4 ops per thread for n=4096
  for (int i = t * 4; i < n; i += 256 * 4) {
    unsigned           lds_off = (unsigned)(uintptr_t)&s[i];   // low 32b of flat = LDS byte offset
    unsigned long long gaddr   = (unsigned long long)(uintptr_t)(d + i);
    asm volatile("global_load_async_to_lds_b128 %0, %1, off"
                 :: "v"(lds_off), "v"(gaddr) : "memory");
  }
  asm volatile("s_wait_asynccnt 0x0" ::: "memory");
  __syncthreads();
  for (int it = 0; it < k; ++it) {
    float best = 3.4e38f; int bi = 0;
    for (int i = t; i < n; i += 256) { float v = s[i]; if (v < best) { best = v; bi = i; } }
    rv[t] = best; ri[t] = bi; __syncthreads();
    for (int st = 128; st > 0; st >>= 1) {
      if (t < st && rv[t + st] < rv[t]) { rv[t] = rv[t + st]; ri[t] = ri[t + st]; }
      __syncthreads();
    }
    if (t == 0) { idx[(size_t)row * k + it] = ri[0]; s[ri[0]] = 3.4e38f; }
    __syncthreads();
  }
}

// ------------------------- edge feature gather -----------------------------
// e[r, :] = [ f_i (C) | f_j - f_i (C) | 0 pad ] in bf16; r = ((b*P+i)*K + s)
__global__ void k_gather(const float* __restrict__ f, int ldf, int C,
                         const int* __restrict__ idx, int Kn,
                         unsigned short* __restrict__ e) {
  int r = blockIdx.x, c = threadIdx.x, Cp2 = blockDim.x;
  int b = r / (PTS * Kn);
  int pi = (r / Kn) % PTS;
  int j = idx[r];
  const float* fi = f + (size_t)(b * PTS + pi) * ldf;
  const float* fj = f + (size_t)(b * PTS + j) * ldf;
  float v = 0.f;
  if (c < C)           v = fi[c];
  else if (c < 2 * C)  v = fj[c - C] - fi[c - C];
  e[(size_t)r * Cp2 + c] = f2bf(v);
}

// ----------------------- layernorm (+relu) fused out -----------------------
// one wave per row (8 rows/block); optional f32 and bf16(zero-padded) outputs
__global__ __launch_bounds__(256) void k_lnorm(const float* __restrict__ in, int C,
                                               const float* __restrict__ g, const float* __restrict__ b,
                                               float* __restrict__ outf,
                                               unsigned short* __restrict__ outbf, int ldbf,
                                               int relu, int rows) {
  int lane = threadIdx.x & 31;
  size_t row = (size_t)blockIdx.x * 8 + (threadIdx.x >> 5);
  if (row >= (size_t)rows) return;
  const float* x = in + row * C;
  float s = 0.f, s2 = 0.f;
  for (int c = lane; c < C; c += 32) { float v = x[c]; s += v; s2 += v * v; }
  s = wred(s); s2 = wred(s2);
  float mean = s / C, var = s2 / C - mean * mean;
  float inv = rsqrtf(var + 1e-5f);
  for (int c = lane; c < C; c += 32) {
    float y = (x[c] - mean) * inv * g[c] + b[c];
    if (relu) y = fmaxf(y, 0.f);
    if (outf)  outf[row * C + c] = y;
    if (outbf) outbf[row * ldbf + c] = f2bf(y);
  }
  if (outbf) for (int c = C + lane; c < ldbf; c += 32) outbf[row * ldbf + c] = 0;
}

// -------------------- max over K neighbors + residual ----------------------
__global__ __launch_bounds__(256) void k_maxaggr(const float* __restrict__ h, int C, int Kn,
                                                 const float* __restrict__ res, int ldres,
                                                 float* __restrict__ out, int ldout, int rows) {
  long long tid = (long long)blockIdx.x * 256 + threadIdx.x;
  if (tid >= (long long)rows * C) return;
  int r = (int)(tid / C), c = (int)(tid % C);
  float m = -3.4e38f;
  const float* hp = h + ((size_t)r * Kn) * C + c;
  for (int s = 0; s < Kn; ++s) m = fmaxf(m, hp[(size_t)s * C]);
  out[(size_t)r * ldout + c] = m + res[(size_t)r * ldres + c];
}

// --------------------------- global max/mean pool --------------------------
__global__ __launch_bounds__(384) void k_gpool(const float* __restrict__ local,
                                               float* __restrict__ gmax, float* __restrict__ gmean) {
  int b = blockIdx.x, c = threadIdx.x;
  float mx = -3.4e38f, sm = 0.f;
  const float* p = local + ((size_t)b * PTS) * 384 + c;
  for (int i = 0; i < PTS; ++i) { float v = p[(size_t)i * 384]; mx = fmaxf(mx, v); sm += v; }
  gmax[b * 384 + c] = mx;
  gmean[b * 384 + c] = sm / (float)PTS;
}

__device__ __forceinline__ float blk_sum512(float v, float* red, int t) {
  red[t] = v; __syncthreads();
  for (int s = 256; s > 0; s >>= 1) { if (t < s) red[t] += red[t + s]; __syncthreads(); }
  float r = red[0]; __syncthreads();
  return r;
}

// global MLP: [768]->512 LN relu ->512 LN relu   (B=4 rows only -> VALU)
__global__ __launch_bounds__(512) void k_gmlp(const float* __restrict__ gmax, const float* __restrict__ gmean,
                                              const float* __restrict__ w1, const float* __restrict__ b1,
                                              const float* __restrict__ n1g, const float* __restrict__ n1b,
                                              const float* __restrict__ w2, const float* __restrict__ b2,
                                              const float* __restrict__ n2g, const float* __restrict__ n2b,
                                              float* __restrict__ gout) {
  __shared__ float h[512];
  __shared__ float red[512];
  int b = blockIdx.x, t = threadIdx.x;
  const float* gm = gmax + b * 384;
  const float* ga = gmean + b * 384;
  const float* wr = w1 + (size_t)t * 768;
  float acc = b1[t];
  for (int i = 0; i < 384; ++i) acc += gm[i] * wr[i];
  for (int i = 0; i < 384; ++i) acc += ga[i] * wr[384 + i];
  float mean = blk_sum512(acc, red, t) * (1.f / 512.f);
  float d = acc - mean;
  float var = blk_sum512(d * d, red, t) * (1.f / 512.f);
  float y = fmaxf(d * rsqrtf(var + 1e-5f) * n1g[t] + n1b[t], 0.f);
  h[t] = y; __syncthreads();
  const float* wr2 = w2 + (size_t)t * 512;
  float acc2 = b2[t];
  for (int i = 0; i < 512; ++i) acc2 += h[i] * wr2[i];
  mean = blk_sum512(acc2, red, t) * (1.f / 512.f);
  d = acc2 - mean;
  var = blk_sum512(d * d, red, t) * (1.f / 512.f);
  gout[b * 512 + t] = fmaxf(d * rsqrtf(var + 1e-5f) * n2g[t] + n2b[t], 0.f);
}

// neighborhood variance: var[r,c] = mean_s (local[j_s,c]-local[r,c])^2
__global__ __launch_bounds__(256) void k_var(const float* __restrict__ local,
                                             const int* __restrict__ bidx,
                                             float* __restrict__ var) {
  long long tid = (long long)blockIdx.x * 256 + threadIdx.x;
  if (tid >= (long long)NPT * 384) return;
  int r = (int)(tid / 384), c = (int)(tid % 384);
  int b = r >> 12;
  float x = local[(size_t)r * 384 + c];
  float acc = 0.f;
  for (int s = 0; s < KBND; ++s) {
    int j = bidx[(size_t)r * KBND + s];
    float d = local[(size_t)(b * PTS + j) * 384 + c] - x;
    acc += d * d;
  }
  var[(size_t)r * 384 + c] = acc * (1.f / KBND);
}

// boundary attention: score = sigmoid(W2 relu(W1 var + b1) + b2)
__global__ __launch_bounds__(64) void k_battn(const float* __restrict__ var,
                                              const float* __restrict__ w1, const float* __restrict__ b1,
                                              const float* __restrict__ w2, const float* __restrict__ b2,
                                              float* __restrict__ score) {
  __shared__ float h[64];
  int r = blockIdx.x, t = threadIdx.x;
  const float* vr = var + (size_t)r * 384;
  const float* wr = w1 + (size_t)t * 384;
  float acc = b1[t];
  for (int i = 0; i < 384; ++i) acc += vr[i] * wr[i];
  h[t] = fmaxf(acc, 0.f);
  __syncthreads();
  if (t == 0) {
    float s = b2[0];
    for (int i = 0; i < 64; ++i) s += h[i] * w2[i];
    score[r] = 1.f / (1.f + expf(-s));
  }
}

// comb = [local(384) | g(512) | score(1) | pad..928] -> bf16
__global__ __launch_bounds__(928) void k_comb(const float* __restrict__ local,
                                              const float* __restrict__ g,
                                              const float* __restrict__ score,
                                              unsigned short* __restrict__ comb) {
  int r = blockIdx.x, c = threadIdx.x;
  int b = r >> 12;
  float v = 0.f;
  if (c < 384)      v = local[(size_t)r * 384 + c];
  else if (c < 896) v = g[b * 512 + (c - 384)];
  else if (c == 896) v = score[r];
  comb[(size_t)r * 928 + c] = f2bf(v);
}

// ArcFace eval: out = clip(cos(emb, w_c), -1, 1) * 16
__global__ __launch_bounds__(256) void k_arc(const float* __restrict__ emb,
                                             const float* __restrict__ aw,
                                             float* __restrict__ out) {
  int lane = threadIdx.x & 31;
  size_t row = (size_t)blockIdx.x * 8 + (threadIdx.x >> 5);
  if (row >= NPT) return;
  const float* e = emb + row * 128;
  float s2 = 0.f;
  for (int c = lane; c < 128; c += 32) s2 += e[c] * e[c];
  s2 = wred(s2);
  float invE = rsqrtf(s2 + 1e-12f);
  for (int cc = 0; cc < 3; ++cc) {
    float dp = 0.f, w2 = 0.f;
    const float* w = aw + cc * 128;
    for (int c = lane; c < 128; c += 32) { float wv = w[c]; dp += e[c] * wv; w2 += wv * wv; }
    dp = wred(dp); w2 = wred(w2);
    if (lane == 0) {
      float cv = dp * invE * rsqrtf(w2 + 1e-12f);
      cv = fminf(1.f, fmaxf(-1.f, cv));
      out[row * 3 + cc] = cv * 16.f;
    }
  }
}

// ------------------------------ host side ----------------------------------

struct Ws {
  unsigned short* fbf;   // [NPT,128] bf16 knn/res features
  float* d2;             // [NPT]
  float* dist;           // [PTS,PTS] per-graph (reused)
  int* idxE;             // [NPT,KEDGE]
  int* idxB;             // [NPT,KBND]
  float* local;          // [NPT,384] (x1|x2|x3|x4)
  float* resb;           // [NPT,128] residual / final emb (aliased)
  unsigned short* edge;  // [NPT*KEDGE,256] bf16 (also comb / e2bf alias)
  float* h1;             // [NPT*KEDGE,128] (also e1 alias)
  unsigned short* h1bf;  // (also e1bf alias)
  float* h2;             // (also e2 alias)
  unsigned short* wbf;   // weight bf16 scratch
  float* gmax; float* gmean; float* gvec;
  float* var; float* score;
};

static void launch_gemm(hipStream_t st, const unsigned short* A, int lda,
                        const unsigned short* Wb, int ldb, const float* bias,
                        float* out, int ldc, int M, int N, int Kpad) {
  dim3 g(M / 128, N / 64), blk(256);
  switch (Kpad) {
    case 32:  k_gemm<32 ><<<g, blk, 0, st>>>(A, lda, Wb, ldb, bias, out, ldc, M); break;
    case 64:  k_gemm<64 ><<<g, blk, 0, st>>>(A, lda, Wb, ldb, bias, out, ldc, M); break;
    case 128: k_gemm<128><<<g, blk, 0, st>>>(A, lda, Wb, ldb, bias, out, ldc, M); break;
    case 256: k_gemm<256><<<g, blk, 0, st>>>(A, lda, Wb, ldb, bias, out, ldc, M); break;
    case 512: k_gemm<512><<<g, blk, 0, st>>>(A, lda, Wb, ldb, bias, out, ldc, M); break;
    case 928: k_gemm<928><<<g, blk, 0, st>>>(A, lda, Wb, ldb, bias, out, ldc, M); break;
    default: break;  // all call sites use the cases above
  }
}

static void run_gemm(hipStream_t st, const Ws& W, const unsigned short* A, int lda,
                     const float* wf, int Kreal, int Kpad, int N,
                     const float* bias, float* out, int M) {
  long long tw = (long long)N * Kpad;
  k_cvtw<<<dim3((unsigned)((tw + 255) / 256)), dim3(256), 0, st>>>(wf, Kreal, Kpad, W.wbf, N);
  launch_gemm(st, A, lda, W.wbf, Kpad, bias, out, N, M, N, Kpad);
}

static void run_knn(hipStream_t st, const Ws& W, const float* fin, int ldf, int C, int pad,
                    int* idxout, int kk) {
  k_cvt<<<dim3((NPT * pad) / 256), dim3(256), 0, st>>>(fin, ldf, C, W.fbf, pad, NPT);
  k_sqnorm<<<dim3(NPT / 256), dim3(256), 0, st>>>(W.fbf, pad, C, W.d2, NPT);
  dim3 g(PTS / 128, PTS / 64), blk(256);
  for (int b = 0; b < BATCH; ++b) {
    const unsigned short* Fb = W.fbf + (size_t)b * PTS * pad;
    const float* d2b = W.d2 + b * PTS;
    if (pad == 32)       k_dist<32 ><<<g, blk, 0, st>>>(Fb, pad, d2b, W.dist);
    else if (pad == 64)  k_dist<64 ><<<g, blk, 0, st>>>(Fb, pad, d2b, W.dist);
    else                 k_dist<128><<<g, blk, 0, st>>>(Fb, pad, d2b, W.dist);
    k_topk<<<dim3(PTS), dim3(256), 0, st>>>(W.dist, idxout + (size_t)b * PTS * kk, PTS, kk);
  }
}

static void run_conv(hipStream_t st, const Ws& W,
                     const float* fin, int ldf, int C, int pad, int Cp2, int Cout,
                     const float* w1, const float* b1, const float* n1g, const float* n1b,
                     const float* w2, const float* b2, const float* n2g, const float* n2b,
                     const float* wres, const float* bres,
                     const float* rsrc, int ldr, float* xout) {
  run_knn(st, W, fin, ldf, C, pad, W.idxE, KEDGE);
  const int rowsE = NPT * KEDGE;
  k_gather<<<dim3(rowsE), dim3(Cp2), 0, st>>>(fin, ldf, C, W.idxE, KEDGE, W.edge);
  run_gemm(st, W, W.edge, Cp2, w1, 2 * C, Cp2, Cout, b1, W.h1, rowsE);
  k_lnorm<<<dim3(rowsE / 8), dim3(256), 0, st>>>(W.h1, Cout, n1g, n1b, nullptr, W.h1bf, Cout, 1, rowsE);
  run_gemm(st, W, W.h1bf, Cout, w2, Cout, Cout, Cout, b2, W.h2, rowsE);
  k_lnorm<<<dim3(rowsE / 8), dim3(256), 0, st>>>(W.h2, Cout, n2g, n2b, W.h2, nullptr, 0, 1, rowsE);
  const float* rptr; int ldrr;
  if (wres) {  // fbf still holds this conv's input features (bf16, zero-padded)
    run_gemm(st, W, W.fbf, pad, wres, C, pad, Cout, bres, W.resb, NPT);
    rptr = W.resb; ldrr = Cout;
  } else { rptr = rsrc; ldrr = ldr; }
  k_maxaggr<<<dim3((NPT * Cout) / 256), dim3(256), 0, st>>>(W.h2, Cout, KEDGE, rptr, ldrr, xout, 384, NPT);
}

extern "C" void kernel_launch(void* const* d_in, const int* in_sizes, int n_in,
                              void* d_out, int out_size, void* d_ws, size_t ws_size,
                              hipStream_t stream) {
  (void)in_sizes; (void)n_in; (void)out_size; (void)ws_size;
  auto F = [&](int i) { return (const float*)d_in[i]; };
  // input order (jax dict sorted): 0:x 1:pos 2:batch 3:arc_w 4..7:battn(l1.b,l1.w,l2.b,l2.w)
  // 8..17:conv1(l1.b,l1.w,l2.b,l2.w,n1.b,n1.g,n2.b,n2.g,res.b,res.w)
  // 18..25:conv2  26..35:conv3  36..43:conv4
  // 44..55:emb(l1.b,l1.w,l2.b,l2.w,l3.b,l3.w,n1.b,n1.g,n2.b,n2.g,n3.b,n3.g)
  // 56..63:gmlp(l1.b,l1.w,l2.b,l2.w,n1.b,n1.g,n2.b,n2.g)
  const float* x   = F(0);
  const float* pos = F(1);

  // ---- workspace carve-out ----
  char* base = (char*)d_ws;
  size_t off = 0;
  auto alloc = [&](size_t bytes) { size_t r = off; off += (bytes + 255) & ~(size_t)255; return (void*)(base + r); };
  Ws W;
  W.fbf   = (unsigned short*)alloc((size_t)NPT * 128 * 2);
  W.d2    = (float*)alloc((size_t)NPT * 4);
  W.dist  = (float*)alloc((size_t)PTS * PTS * 4);
  W.idxE  = (int*)alloc((size_t)NPT * KEDGE * 4);
  W.idxB  = (int*)alloc((size_t)NPT * KBND * 4);
  W.local = (float*)alloc((size_t)NPT * 384 * 4);
  W.resb  = (float*)alloc((size_t)NPT * 128 * 4);
  W.edge  = (unsigned short*)alloc((size_t)NPT * KEDGE * 256 * 2);
  W.h1    = (float*)alloc((size_t)NPT * KEDGE * 128 * 4);
  W.h1bf  = (unsigned short*)alloc((size_t)NPT * KEDGE * 128 * 2);
  W.h2    = (float*)alloc((size_t)NPT * KEDGE * 128 * 4);
  W.wbf   = (unsigned short*)alloc((size_t)512 * 928 * 2);
  W.gmax  = (float*)alloc((size_t)BATCH * 384 * 4);
  W.gmean = (float*)alloc((size_t)BATCH * 384 * 4);
  W.gvec  = (float*)alloc((size_t)BATCH * 512 * 4);
  W.var   = (float*)alloc((size_t)NPT * 384 * 4);
  W.score = (float*)alloc((size_t)NPT * 4);

  float* x1 = W.local + 0;
  float* x2 = W.local + 64;
  float* x3 = W.local + 128;
  float* x4 = W.local + 256;

  // ---- edge convs ----
  run_conv(stream, W, x, 6, 6, 32, 32, 64,
           F(9), F(8), F(13), F(12), F(11), F(10), F(15), F(14),
           F(17), F(16), nullptr, 0, x1);
  run_conv(stream, W, x1, 384, 64, 64, 128, 64,
           F(19), F(18), F(23), F(22), F(21), F(20), F(25), F(24),
           nullptr, nullptr, x1, 384, x2);
  run_conv(stream, W, x2, 384, 64, 64, 128, 128,
           F(27), F(26), F(31), F(30), F(29), F(28), F(33), F(32),
           F(35), F(34), nullptr, 0, x3);
  run_conv(stream, W, x3, 384, 128, 128, 256, 128,
           F(37), F(36), F(41), F(40), F(39), F(38), F(43), F(42),
           nullptr, nullptr, x3, 384, x4);

  // ---- global pooling + MLP ----
  k_gpool<<<dim3(BATCH), dim3(384), 0, stream>>>(W.local, W.gmax, W.gmean);
  k_gmlp<<<dim3(BATCH), dim3(512), 0, stream>>>(W.gmax, W.gmean,
      F(57), F(56), F(61), F(60), F(59), F(58), F(63), F(62), W.gvec);

  // ---- boundary attention ----
  run_knn(stream, W, pos, 3, 3, 32, W.idxB, KBND);
  k_var<<<dim3((NPT * 384) / 256), dim3(256), 0, stream>>>(W.local, W.idxB, W.var);
  k_battn<<<dim3(NPT), dim3(64), 0, stream>>>(W.var, F(5), F(4), F(7), F(6), W.score);

  // ---- embedding MLP (comb/e* alias the big edge buffers, now dead) ----
  unsigned short* comb = W.edge;            // [NPT,928] bf16
  float* e1   = W.h1;                       // [NPT,512]
  unsigned short* e1bf = W.h1bf;            // [NPT,512]
  float* e2   = W.h2;                       // [NPT,256]
  unsigned short* e2bf = W.edge;            // [NPT,256] (comb dead after l1 gemm)
  float* embf = W.resb;                     // [NPT,128]

  k_comb<<<dim3(NPT), dim3(928), 0, stream>>>(W.local, W.gvec, W.score, comb);
  run_gemm(stream, W, comb, 928, F(45), 897, 928, 512, F(44), e1, NPT);
  k_lnorm<<<dim3(NPT / 8), dim3(256), 0, stream>>>(e1, 512, F(51), F(50), nullptr, e1bf, 512, 1, NPT);
  run_gemm(stream, W, e1bf, 512, F(47), 512, 512, 256, F(46), e2, NPT);
  k_lnorm<<<dim3(NPT / 8), dim3(256), 0, stream>>>(e2, 256, F(53), F(52), nullptr, e2bf, 256, 1, NPT);
  run_gemm(stream, W, e2bf, 256, F(49), 256, 256, 128, F(48), embf, NPT);
  k_lnorm<<<dim3(NPT / 8), dim3(256), 0, stream>>>(embf, 128, F(55), F(54), embf, nullptr, 0, 0, NPT);

  // ---- ArcFace head ----
  k_arc<<<dim3(NPT / 8), dim3(256), 0, stream>>>(embf, F(3), (float*)d_out);
}
